// AttentionLayer_67817533604105
// MI455X (gfx1250) — matrix operand order repaired and checked by reading, not compiled, not run on hardware
//
#include <hip/hip_runtime.h>
#include <hip/hip_bf16.h>

// Problem constants (match reference)
#define B_    4
#define S_    4096
#define DIN_  256
#define DOUT_ 64
#define BS_   (B_ * S_)

typedef __attribute__((ext_vector_type(16))) _Float16     v16h;
typedef __attribute__((ext_vector_type(8)))  float        v8f;
typedef __attribute__((ext_vector_type(4)))  unsigned int v4u;
typedef __attribute__((ext_vector_type(2)))  unsigned int v2u;

union HFrag {
    v16h     v;
    _Float16 h[16];
    v4u      q[2];
};
union Pack8 { _Float16 h[8]; v4u q; };
union Pack4 { _Float16 h[4]; v2u q; };

static __device__ __forceinline__ v8f wmma_f16(v16h a, v16h b, v8f c) {
    // D = A(16x32 f16) * B(32x16 f16) + C(16x16 f32)
    return __builtin_amdgcn_wmma_f32_16x16x32_f16(
        /*neg_a=*/false, a, /*neg_b=*/false, b,
        /*c_mod=*/(short)0, c, /*reuse_a=*/false, /*reuse_b=*/false);
}

// Q pre-scale: 1/sqrt(DOUT) * log2(e)  -> softmax runs in base-2 domain,
// exponentials become raw v_exp_f32 (natively base-2) with no extra mul.
#define QSCALE (0.125f * 1.44269504088896340736f)

// ---------------------------------------------------------------------------
// Kernel 1: fused QKV projection.  grid = (BS/64, 3), block = 128 (4 waves).
// Workgroup owns 64 rows of x (staged once in LDS as f16); wave w owns output
// cols [16w,16w+16) and preloads all 8 W B-fragments into registers, then
// runs 4 m-tiles x 8 WMMAs.  blockIdx.y selects {Q,K,V}.
//   Q: row-major f16, pre-scaled by QSCALE
//   K: row-major f16
//   V: TRANSPOSED f16  [B][DOUT][S]  (one b128 store per lane per m-tile)
// ---------------------------------------------------------------------------
__global__ void qkv_proj_kernel(const float* __restrict__ x,
                                const float* __restrict__ Wq, const float* __restrict__ bq,
                                const float* __restrict__ Wk, const float* __restrict__ bk,
                                const float* __restrict__ Wv, const float* __restrict__ bv,
                                _Float16* __restrict__ qw,
                                _Float16* __restrict__ kw,
                                _Float16* __restrict__ vwT)
{
    const int tile  = blockIdx.x;   // 64-row tile over flattened B*S rows
    const int which = blockIdx.y;   // 0=Q, 1=K, 2=V

    const float* W;  const float* bias;
    if (which == 0)      { W = Wq; bias = bq; }
    else if (which == 1) { W = Wk; bias = bk; }
    else                 { W = Wv; bias = bv; }

    __shared__ __align__(16) _Float16 lx[64 * DIN_];   // 32 KB x tile in f16

    const int R0 = tile * 64;
    {   // vectorized f32 -> f16 staging (float4 in, 8B out)
        const float4* xs = (const float4*)(x + (size_t)R0 * DIN_);
        for (int i = threadIdx.x; i < (64 * DIN_) / 4; i += 128) {
            float4 f = xs[i];
            Pack4 p;
            p.h[0] = (_Float16)f.x; p.h[1] = (_Float16)f.y;
            p.h[2] = (_Float16)f.z; p.h[3] = (_Float16)f.w;
            *(v2u*)&lx[i * 4] = p.q;
        }
    }
    __syncthreads();

    const int lane = threadIdx.x & 31;
    const int w    = threadIdx.x >> 5;   // N-tile 0..3
    const int g    = lane >> 4;          // half-wave group
    const int lr   = lane & 15;

    // Preload all 8 W B-fragments for this wave's N-tile (K = 0..255)
    HFrag wf[8];
#pragma unroll
    for (int k0i = 0; k0i < 8; ++k0i) {
#pragma unroll
        for (int h = 0; h < 16; ++h)
            wf[k0i].h[h] =
                (_Float16)W[(size_t)(k0i * 32 + h + 16 * g) * DOUT_ + w * 16 + lr];
    }
    const float bval = bias[w * 16 + lr];
    const float osc  = (which == 0) ? QSCALE : 1.0f;

#pragma unroll
    for (int mt = 0; mt < 4; ++mt) {
        v8f c = {};
#pragma unroll
        for (int k0i = 0; k0i < 8; ++k0i) {
            HFrag a;   // lane = row lr; halves 0..15 -> K=16g..16g+15
            a.q[0] = *(const v4u*)&lx[(mt * 16 + lr) * DIN_ + k0i * 32 + 16 * g];
            a.q[1] = *(const v4u*)&lx[(mt * 16 + lr) * DIN_ + k0i * 32 + 16 * g + 8];
            c = wmma_f16(a.v, wf[k0i].v, c);
        }
        const int r0m = R0 + mt * 16;
        if (which < 2) {
            _Float16* out = (which == 0) ? qw : kw;
#pragma unroll
            for (int j = 0; j < 8; ++j)    // C layout: row = j + 8g, col = 16w+lr
                out[(size_t)(r0m + j + 8 * g) * DOUT_ + w * 16 + lr] =
                    (_Float16)((c[j] + bval) * osc);
        } else {
            // V transposed: lane holds 8 consecutive rows of one column
            const int bb = r0m / S_;
            const int s0 = (r0m % S_) + 8 * g;
            Pack8 p;
#pragma unroll
            for (int j = 0; j < 8; ++j) p.h[j] = (_Float16)(c[j] + bval);
            *(v4u*)&vwT[(size_t)bb * DOUT_ * S_ + (size_t)(w * 16 + lr) * S_ + s0] = p.q;
        }
    }
}

// ---------------------------------------------------------------------------
// Kernel 2: flash attention.  grid = B * S/64, block = 128 (4 independent
// waves, no barriers).  Wave owns 16 query rows; keys in 64-wide tiles.
// Score tile t, column n is assigned key (4n+t) so each lane's four P values
// are CONTIGUOUS in the P K-index -> one b64 LDS store per row (softmax order
// is key-order invariant; V^T / A-fragment indexing stays natural).
// Uniform branch skips the O/l rescale when the running max is unchanged.
// ---------------------------------------------------------------------------
__global__ void flash_attn_kernel(const _Float16* __restrict__ qw,
                                  const _Float16* __restrict__ kw,
                                  const _Float16* __restrict__ vwT,
                                  float* __restrict__ zout)
{
    const int blk  = blockIdx.x;
    const int b    = blk >> 6;           // / (S/64)
    const int mblk = blk & 63;
    const int lane = threadIdx.x & 31;
    const int w    = threadIdx.x >> 5;
    const int g    = lane >> 4;
    const int lr   = lane & 15;

    __shared__ __align__(16) _Float16 lp[4][16 * 64];   // P staging, 2KB/wave

    const size_t    base = (size_t)b * S_ * DOUT_;
    const _Float16* vt   = vwT + (size_t)b * DOUT_ * S_;
    const int       m0   = mblk * 64 + w * 16;

    // Q fragments (pre-scaled by QSCALE): two 16x32 A fragments
    HFrag aq[2];
    {
        const _Float16* qrow = qw + base + (size_t)(m0 + lr) * DOUT_;
#pragma unroll
        for (int k0i = 0; k0i < 2; ++k0i) {
            aq[k0i].q[0] = *(const v4u*)(qrow + k0i * 32 + 16 * g);
            aq[k0i].q[1] = *(const v4u*)(qrow + k0i * 32 + 16 * g + 8);
        }
    }

    float mstate[8], lstate[8];
#pragma unroll
    for (int j = 0; j < 8; ++j) { mstate[j] = -1e30f; lstate[j] = 0.0f; }
    v8f acc[4] = {};

    for (int n0 = 0; n0 < S_; n0 += 64) {
        const _Float16* kt = kw + base + (size_t)n0 * DOUT_;

        // Prefetch next iteration's K tile (8KB) and V^T stripes
        if (n0 + 64 < S_) {
            __builtin_prefetch(kt + 64 * DOUT_ + lane * 128, 0, 1);
            __builtin_prefetch(kt + 64 * DOUT_ + lane * 128 + 64, 0, 1);
            __builtin_prefetch(vt + (size_t)(lane * 2) * S_ + n0 + 64, 0, 1);
            __builtin_prefetch(vt + (size_t)(lane * 2 + 1) * S_ + n0 + 64, 0, 1);
        }

        // ---- scores: tile t, col n <- key 4n+t (four 16x16 C tiles) ----
        v8f c[4] = {};
#pragma unroll
        for (int k0i = 0; k0i < 2; ++k0i) {
            const int k0 = k0i * 32;
#pragma unroll
            for (int t = 0; t < 4; ++t) {
                // B fragment: col lr -> key row (4*lr + t); half h -> d = k0+h+16g
                const _Float16* kp = kt + (size_t)(4 * lr + t) * DOUT_ + k0 + 16 * g;
                HFrag bk;
                bk.q[0] = *(const v4u*)kp;
                bk.q[1] = *(const v4u*)(kp + 8);
                c[t] = wmma_f16(aq[k0i].v, bk.v, c[t]);
            }
        }

        // ---- online softmax in base-2 (row = j + 8g; 64 keys) ----
        float mnew[8];
        bool changed = false;
#pragma unroll
        for (int j = 0; j < 8; ++j) {
            float rmax = fmaxf(fmaxf(c[0][j], c[1][j]), fmaxf(c[2][j], c[3][j]));
#pragma unroll
            for (int off = 1; off < 16; off <<= 1)
                rmax = fmaxf(rmax, __shfl_xor(rmax, off, 32));
            mnew[j] = fmaxf(mstate[j], rmax);
            changed |= (mnew[j] > mstate[j]);
        }
        if (__any(changed)) {   // uniform branch: rescale O and l
#pragma unroll
            for (int j = 0; j < 8; ++j) {
                const float corr = __builtin_amdgcn_exp2f(mstate[j] - mnew[j]);
                lstate[j] *= corr;
                mstate[j] = mnew[j];
#pragma unroll
                for (int nt = 0; nt < 4; ++nt) acc[nt][j] *= corr;
            }
        }
#pragma unroll
        for (int j = 0; j < 8; ++j) {
            Pack4 p;
            float pf[4];
#pragma unroll
            for (int t = 0; t < 4; ++t) {
                pf[t] = __builtin_amdgcn_exp2f(c[t][j] - mstate[j]);
                p.h[t] = (_Float16)pf[t];
            }
            float rsum = (pf[0] + pf[1]) + (pf[2] + pf[3]);
#pragma unroll
            for (int off = 1; off < 16; off <<= 1)
                rsum += __shfl_xor(rsum, off, 32);
            lstate[j] += rsum;
            // lane's keys 4lr..4lr+3 are contiguous -> single b64 store
            const int prow = j + 8 * g;
            *(v2u*)&lp[w][prow * 64 + 4 * lr] = p.q;
        }
        // DS ops from one wave complete in order; stop compiler reordering only.
        __asm__ volatile("" ::: "memory");

        // ---- O += P * V : P is 16x64 -> two A fragments (kb = K-block) ----
#pragma unroll
        for (int kb = 0; kb < 2; ++kb) {
            HFrag ap;
            ap.q[0] = *(const v4u*)&lp[w][lr * 64 + kb * 32 + 16 * g];
            ap.q[1] = *(const v4u*)&lp[w][lr * 64 + kb * 32 + 16 * g + 8];
#pragma unroll
            for (int nt = 0; nt < 4; ++nt) {
                // B fragment from V^T: row d = nt*16+lr; half h -> key n0+kb*32+h+16g
                const _Float16* vp =
                    vt + (size_t)(nt * 16 + lr) * S_ + n0 + kb * 32 + 16 * g;
                HFrag bv;
                bv.q[0] = *(const v4u*)vp;
                bv.q[1] = *(const v4u*)(vp + 8);
                acc[nt] = wmma_f16(ap.v, bv.v, acc[nt]);
            }
        }
    }

    // ---- epilogue: divide by row sums, store f32 ----
#pragma unroll
    for (int j = 0; j < 8; ++j) {
        const float inv = __builtin_amdgcn_rcpf(lstate[j]);
        const int row = m0 + j + 8 * g;
        float* orow = zout + ((size_t)b * S_ + row) * DOUT_;
#pragma unroll
        for (int nt = 0; nt < 4; ++nt)
            orow[nt * 16 + lr] = acc[nt][j] * inv;
    }
}

// ---------------------------------------------------------------------------
extern "C" void kernel_launch(void* const* d_in, const int* in_sizes, int n_in,
                              void* d_out, int out_size, void* d_ws, size_t ws_size,
                              hipStream_t stream)
{
    const float* x  = (const float*)d_in[0];
    const float* Wq = (const float*)d_in[1];
    const float* bq = (const float*)d_in[2];
    const float* Wk = (const float*)d_in[3];
    const float* bk = (const float*)d_in[4];
    const float* Wv = (const float*)d_in[5];
    const float* bv = (const float*)d_in[6];
    float* z = (float*)d_out;

    // Workspace: q (row-major, pre-scaled), k (row-major), v^T -- 2 MB each
    const size_t per = (size_t)BS_ * DOUT_;
    _Float16* qw  = (_Float16*)d_ws;
    _Float16* kw  = qw + per;
    _Float16* vwT = kw + per;

    dim3 g1(BS_ / 64, 3);
    qkv_proj_kernel<<<g1, 128, 0, stream>>>(x, Wq, bq, Wk, bk, Wv, bv, qw, kw, vwT);

    flash_attn_kernel<<<B_ * (S_ / 64), 128, 0, stream>>>(qw, kw, vwT, z);
}
